// RelPositionMultiHeadAttention_13271448944920
// MI455X (gfx1250) — matrix operand order repaired and checked by reading, not compiled, not run on hardware
//
#include <hip/hip_runtime.h>

typedef __attribute__((ext_vector_type(16))) __bf16 v16bf;
typedef __attribute__((ext_vector_type(8)))  __bf16 v8bf;
typedef __attribute__((ext_vector_type(8)))  float  v8f;

#define DEV static __device__ __forceinline__

// problem dims
constexpr int BB = 4, TT = 2048, EE = 512, HH = 8, DD = 64;
constexpr int HD = HH * DD;          // 512
constexpr int N3 = 3 * HD;           // 1536
constexpr float ATT_SCALE = 0.125f;  // 1/sqrt(64)

DEV int laneid() { return (int)(threadIdx.x & 31u); }

DEV v16bf zero16() {
  union { unsigned int u[8]; v16bf v; } z;
#pragma unroll
  for (int i = 0; i < 8; ++i) z.u[i] = 0u;
  return z.v;
}
DEV v8f zero8() { v8f z = {0.f, 0.f, 0.f, 0.f, 0.f, 0.f, 0.f, 0.f}; return z; }

DEV v16bf cat8(v8bf lo, v8bf hi) {
  return __builtin_shufflevector(lo, hi, 0, 1, 2, 3, 4, 5, 6, 7, 8, 9, 10, 11, 12, 13, 14, 15);
}

// ---- WMMA fragment loaders (contiguous k order: element s holds k = 16*(lane/16) + s, wave32) ----
// A 16x32 bf16: lane L<16 -> row L, K 0..15 ; lane L>=16 -> row L-16, K 16..31
DEV v16bf load_a_row(const __bf16* mat, int row, int ld) {
  const int lane = laneid();
  const int kb = (lane >> 4) << 4;
  const __bf16* p = mat + (size_t)row * ld + kb;
  v8bf lo = *(const v8bf*)(p);
  v8bf hi = *(const v8bf*)(p + 8);
  return cat8(lo, hi);
}
DEV v16bf load_a(const __bf16* tile, int ld) { return load_a_row(tile, laneid() & 15, ld); }

// B 32x16 bf16 from B^T (n-major) storage: lane L<16 -> col L, K 0..15 ; lane L>=16 -> col L-16, K 16..31
DEV v16bf load_b(const __bf16* bt, int ld) {
  const int lane = laneid();
  const int n  = lane & 15;
  const int kb = (lane >> 4) << 4;
  const __bf16* p = bt + (size_t)n * ld + kb;
  v8bf lo = *(const v8bf*)(p);
  v8bf hi = *(const v8bf*)(p + 8);
  return cat8(lo, hi);
}
// same, but per-lane row = rowbase + n, rows outside [0,rowmax) read as zero
DEV v16bf load_b_win(const __bf16* mat, int rowbase, int ld, int rowmax) {
  const int lane = laneid();
  const int n   = lane & 15;
  const int row = rowbase + n;
  if (row < 0 || row >= rowmax) return zero16();
  const int kb = (lane >> 4) << 4;
  const __bf16* p = mat + (size_t)row * ld + kb;
  v8bf lo = *(const v8bf*)(p);
  v8bf hi = *(const v8bf*)(p + 8);
  return cat8(lo, hi);
}

DEV v8f wmma_bf16(v16bf a, v16bf b, v8f c) {
  return __builtin_amdgcn_wmma_f32_16x16x32_bf16(false, a, false, b, (short)0, c, false, false);
}

// C/D f32 16x16: element (m,n) -> lane = n + 16*(m/8), vgpr = m%8
DEV void store_c(float* buf, int ld, int col0, v8f c) {
  const int lane = laneid();
  const int nl = lane & 15;
  const int mh = (lane >> 4) << 3;
#pragma unroll
  for (int g = 0; g < 8; ++g) buf[(mh + g) * ld + col0 + nl] = c[g];
}

union F8 { v8f v; float f[8]; };

// Software-pipelined 16(M) x 64(N) x K GEMM macro-tile: loads for step k+1 issued
// before WMMAs of step k so s_wait_loadcnt waits on older loads only.
DEV void gemm16x64(const __bf16* Ab, const __bf16* Bb, int lda, int ldb, int K,
                   v8f& c0, v8f& c1, v8f& c2, v8f& c3) {
  v16bf a  = load_a(Ab, lda);
  v16bf b0 = load_b(Bb + (size_t) 0 * ldb, ldb);
  v16bf b1 = load_b(Bb + (size_t)16 * ldb, ldb);
  v16bf b2 = load_b(Bb + (size_t)32 * ldb, ldb);
  v16bf b3 = load_b(Bb + (size_t)48 * ldb, ldb);
  for (int k0 = 32; k0 < K; k0 += 32) {
    v16bf an  = load_a(Ab + k0, lda);
    v16bf bn0 = load_b(Bb + (size_t) 0 * ldb + k0, ldb);
    v16bf bn1 = load_b(Bb + (size_t)16 * ldb + k0, ldb);
    v16bf bn2 = load_b(Bb + (size_t)32 * ldb + k0, ldb);
    v16bf bn3 = load_b(Bb + (size_t)48 * ldb + k0, ldb);
    c0 = wmma_bf16(a, b0, c0);
    c1 = wmma_bf16(a, b1, c1);
    c2 = wmma_bf16(a, b2, c2);
    c3 = wmma_bf16(a, b3, c3);
    a = an; b0 = bn0; b1 = bn1; b2 = bn2; b3 = bn3;
  }
  c0 = wmma_bf16(a, b0, c0);
  c1 = wmma_bf16(a, b1, c1);
  c2 = wmma_bf16(a, b2, c2);
  c3 = wmma_bf16(a, b3, c3);
}

// ---------------- prep kernels ----------------
__global__ void k_convert(const float* __restrict__ src, __bf16* __restrict__ dst, int n) {
  int i = blockIdx.x * 256 + threadIdx.x;
  if (i < n) dst[i] = (__bf16)src[i];
}

// dst[n*K + k] = (bf16)src[k*N + n] ; K,N multiples of 16
__global__ void k_transpose(const float* __restrict__ src, __bf16* __restrict__ dst, int K, int N) {
  __shared__ float tile[16][17];
  int k = blockIdx.y * 16 + threadIdx.y;
  int n = blockIdx.x * 16 + threadIdx.x;
  tile[threadIdx.y][threadIdx.x] = src[(size_t)k * N + n];
  __syncthreads();
  int k2 = blockIdx.y * 16 + threadIdx.x;
  int n2 = blockIdx.x * 16 + threadIdx.y;
  dst[(size_t)n2 * K + k2] = (__bf16)tile[threadIdx.x][threadIdx.y];
}

// ---------------- QKV projection: [B*T,E] @ [E,3HD], scatter with biases ----------------
__global__ void __launch_bounds__(32) k_qkv(const __bf16* __restrict__ X, const __bf16* __restrict__ WT,
                                            const float* __restrict__ rwb, const float* __restrict__ rrb,
                                            __bf16* __restrict__ qw, __bf16* __restrict__ qr,
                                            __bf16* __restrict__ kk, __bf16* __restrict__ vT) {
  const int n0 = blockIdx.x * 64;   // over 3HD, never straddles q/k/v or heads
  const int m0 = blockIdx.y * 16;   // over B*T
  v8f a0 = zero8(), a1 = zero8(), a2 = zero8(), a3 = zero8();
  gemm16x64(X + (size_t)m0 * EE, WT + (size_t)n0 * EE, EE, EE, EE, a0, a1, a2, a3);
  const int lane = laneid(), nl = lane & 15, mh = (lane >> 4) << 3;
  const int sec = n0 >> 9;            // 0=q 1=k 2=v
  const int h = (n0 & 511) >> 6;      // uniform per block
  F8 u[4]; u[0].v = a0; u[1].v = a1; u[2].v = a2; u[3].v = a3;
#pragma unroll
  for (int nt = 0; nt < 4; ++nt) {
    const int d = (nt << 4) + nl;
#pragma unroll
    for (int g = 0; g < 8; ++g) {
      const int m = m0 + mh + g;
      const int b = m >> 11;
      const int t = m & 2047;
      const float v = u[nt].f[g];
      if (sec == 0) {
        const size_t idx = (((size_t)b * HH + h) * TT + t) * DD + d;
        qw[idx] = (__bf16)(v + rwb[h * DD + d]);
        qr[idx] = (__bf16)(v + rrb[h * DD + d]);
      } else if (sec == 1) {
        kk[(((size_t)b * HH + h) * TT + t) * DD + d] = (__bf16)v;
      } else {
        vT[(((size_t)b * HH + h) * DD + d) * TT + t] = (__bf16)v;
      }
    }
  }
}

// ---------------- positional projection: r = pos_enc @ w_pos -> rr[h][t][d] ----------------
__global__ void __launch_bounds__(32) k_pos(const __bf16* __restrict__ P, const __bf16* __restrict__ WT,
                                            __bf16* __restrict__ rr) {
  const int n0 = blockIdx.x * 64;   // over HD
  const int m0 = blockIdx.y * 16;   // over T
  v8f a0 = zero8(), a1 = zero8(), a2 = zero8(), a3 = zero8();
  gemm16x64(P + (size_t)m0 * EE, WT + (size_t)n0 * EE, EE, EE, EE, a0, a1, a2, a3);
  const int lane = laneid(), nl = lane & 15, mh = (lane >> 4) << 3;
  const int h = n0 >> 6;
  F8 u[4]; u[0].v = a0; u[1].v = a1; u[2].v = a2; u[3].v = a3;
#pragma unroll
  for (int nt = 0; nt < 4; ++nt) {
    const int d = (nt << 4) + nl;
#pragma unroll
    for (int g = 0; g < 8; ++g) {
      const int t = m0 + mh + g;
      rr[((size_t)h * TT + t) * DD + d] = (__bf16)u[nt].f[g];
    }
  }
}

// ---------------- streaming rel-attention: one wave per (b,h,16 q-rows) ----------------
__global__ void __launch_bounds__(32) k_attn(const __bf16* __restrict__ qw, const __bf16* __restrict__ qr,
                                             const __bf16* __restrict__ kk, const __bf16* __restrict__ vT,
                                             const __bf16* __restrict__ rr, const int* __restrict__ lens,
                                             __bf16* __restrict__ attn) {
  const int i0 = blockIdx.x * 16;
  const int h  = blockIdx.y;
  const int b  = blockIdx.z;
  const size_t bh = (size_t)b * HH + h;
  const __bf16* qw_bh = qw + bh * TT * DD;
  const __bf16* qr_bh = qr + bh * TT * DD;
  const __bf16* kk_bh = kk + bh * TT * DD;
  const __bf16* vT_bh = vT + bh * DD * TT;
  const __bf16* rr_h  = rr + (size_t)h * TT * DD;
  const int len = lens[b];

  __shared__ float  s_ac[16 * 32];
  __shared__ float  s_w1[16 * 48];
  __shared__ float  s_w2[16 * 48];
  __shared__ alignas(16) __bf16 s_p[16 * 32];
  __shared__ float  s_m[16], s_l[16], s_c[16];

  const int lane = laneid();
  const int rl = lane & 15;
  const int mh = (lane >> 4) << 3;
  const int nl = lane & 15;

  // resident A fragments: q+rw_bias rows i0.., q+rr_bias rows i0.., q+rr_bias rows i0+1..
  v16bf qwA0 = load_a(qw_bh + (size_t)i0 * DD +  0, DD);
  v16bf qwA1 = load_a(qw_bh + (size_t)i0 * DD + 32, DD);
  v16bf qrA0 = load_a(qr_bh + (size_t)i0 * DD +  0, DD);
  v16bf qrA1 = load_a(qr_bh + (size_t)i0 * DD + 32, DD);
  int row1 = i0 + 1 + rl; if (row1 > TT - 1) row1 = TT - 1;  // clamped row never consumed
  v16bf q1A0 = load_a_row(qr_bh +  0, row1, DD);
  v16bf q1A1 = load_a_row(qr_bh + 32, row1, DD);

  if (lane < 16) { s_m[lane] = -3.0e38f; s_l[lane] = 0.0f; }
  __syncthreads();

  v8f o0 = zero8(), o1 = zero8(), o2 = zero8(), o3 = zero8();

  const int jmax = ((len + 31) >> 5) << 5;  // chunks fully masked contribute exactly 0
  for (int j0 = 0; j0 < jmax; j0 += 32) {
    if (j0 + 32 < jmax) __builtin_prefetch((const void*)(kk_bh + (size_t)(j0 + 32) * DD), 0, 1);
    // ---- issue all score-stage B fragments up front, then back-to-back WMMAs ----
    const bool needW1 = (j0 <= i0 + 15);
    const bool needW2 = (j0 + 30 > i0);
    const int m1lo = (TT - 1) + (j0 - i0) - 15;
    const int m2lo = (j0 - i0) - 17;

    v16bf kb00 = load_b(kk_bh + (size_t)(j0     ) * DD +  0, DD);
    v16bf kb01 = load_b(kk_bh + (size_t)(j0     ) * DD + 32, DD);
    v16bf kb10 = load_b(kk_bh + (size_t)(j0 + 16) * DD +  0, DD);
    v16bf kb11 = load_b(kk_bh + (size_t)(j0 + 16) * DD + 32, DD);

    // AC = (q + r_w_bias) @ K^T
    v8f ac0 = zero8(), ac1 = zero8();
    ac0 = wmma_bf16(qwA0, kb00, ac0);
    ac0 = wmma_bf16(qwA1, kb01, ac0);
    ac1 = wmma_bf16(qwA0, kb10, ac1);
    ac1 = wmma_bf16(qwA1, kb11, ac1);
    store_c(s_ac, 32, 0, ac0);
    store_c(s_ac, 32, 16, ac1);

    // W1[ii][c] = (q[i0+ii]+rrb) . r[m1lo + c]   (consumed for j<=i)
    if (needW1) {
      v16bf wa0 = load_b_win(rr_h +  0, m1lo +  0, DD, TT);
      v16bf wa1 = load_b_win(rr_h + 32, m1lo +  0, DD, TT);
      v16bf wb0 = load_b_win(rr_h +  0, m1lo + 16, DD, TT);
      v16bf wb1 = load_b_win(rr_h + 32, m1lo + 16, DD, TT);
      v16bf wc0 = load_b_win(rr_h +  0, m1lo + 32, DD, TT);
      v16bf wc1 = load_b_win(rr_h + 32, m1lo + 32, DD, TT);
      v8f w0 = zero8(), w1 = zero8(), w2 = zero8();
      w0 = wmma_bf16(qrA0, wa0, w0);
      w0 = wmma_bf16(qrA1, wa1, w0);
      w1 = wmma_bf16(qrA0, wb0, w1);
      w1 = wmma_bf16(qrA1, wb1, w1);
      w2 = wmma_bf16(qrA0, wc0, w2);
      w2 = wmma_bf16(qrA1, wc1, w2);
      store_c(s_w1, 48,  0, w0);
      store_c(s_w1, 48, 16, w1);
      store_c(s_w1, 48, 32, w2);
    }
    // W2[ii][c] = (q[i0+ii+1]+rrb) . r[m2lo + c]  (consumed for j>i+1)
    if (needW2) {
      v16bf wa0 = load_b_win(rr_h +  0, m2lo +  0, DD, TT);
      v16bf wa1 = load_b_win(rr_h + 32, m2lo +  0, DD, TT);
      v16bf wb0 = load_b_win(rr_h +  0, m2lo + 16, DD, TT);
      v16bf wb1 = load_b_win(rr_h + 32, m2lo + 16, DD, TT);
      v16bf wc0 = load_b_win(rr_h +  0, m2lo + 32, DD, TT);
      v16bf wc1 = load_b_win(rr_h + 32, m2lo + 32, DD, TT);
      v8f w0 = zero8(), w1 = zero8(), w2 = zero8();
      w0 = wmma_bf16(q1A0, wa0, w0);
      w0 = wmma_bf16(q1A1, wa1, w0);
      w1 = wmma_bf16(q1A0, wb0, w1);
      w1 = wmma_bf16(q1A1, wb1, w1);
      w2 = wmma_bf16(q1A0, wc0, w2);
      w2 = wmma_bf16(q1A1, wc1, w2);
      store_c(s_w2, 48,  0, w0);
      store_c(s_w2, 48, 16, w1);
      store_c(s_w2, 48, 32, w2);
    }
    __syncthreads();
    // online softmax, one row per thread 0..15
    if (lane < 16) {
      const int i = i0 + lane;
      const float mprev = s_m[lane];
      float lsum = s_l[lane];
      float sv[32];
      float mc = -3.0e38f;
#pragma unroll
      for (int jj = 0; jj < 32; ++jj) {
        const int j = j0 + jj;
        const int c = jj - lane + 15;  // in [0,46]
        float bd;
        if (j <= i)          bd = s_w1[lane * 48 + c];
        else if (j == i + 1) bd = 0.0f;
        else                 bd = s_w2[lane * 48 + c];
        float s = (s_ac[lane * 32 + jj] + bd) * ATT_SCALE;
        if (j >= len) s = -10000.0f;
        sv[jj] = s;
        mc = fmaxf(mc, s);
      }
      const float mnew = fmaxf(mprev, mc);
      const float corr = __expf(mprev - mnew);
      lsum *= corr;
#pragma unroll
      for (int jj = 0; jj < 32; ++jj) {
        const float p = __expf(sv[jj] - mnew);
        lsum += p;
        s_p[lane * 32 + jj] = (__bf16)p;
      }
      s_m[lane] = mnew; s_l[lane] = lsum; s_c[lane] = corr;
    }
    __syncthreads();
    // rescale accumulators, then O += P @ V
    v16bf vb0 = load_b(vT_bh + (size_t)( 0) * TT + j0, TT);
    v16bf vb1 = load_b(vT_bh + (size_t)(16) * TT + j0, TT);
    v16bf vb2 = load_b(vT_bh + (size_t)(32) * TT + j0, TT);
    v16bf vb3 = load_b(vT_bh + (size_t)(48) * TT + j0, TT);
    float cf[8];
#pragma unroll
    for (int g = 0; g < 8; ++g) cf[g] = s_c[mh + g];
#pragma unroll
    for (int g = 0; g < 8; ++g) {
      o0[g] = o0[g] * cf[g]; o1[g] = o1[g] * cf[g];
      o2[g] = o2[g] * cf[g]; o3[g] = o3[g] * cf[g];
    }
    v16bf pA = load_a((const __bf16*)s_p, 32);
    o0 = wmma_bf16(pA, vb0, o0);
    o1 = wmma_bf16(pA, vb1, o1);
    o2 = wmma_bf16(pA, vb2, o2);
    o3 = wmma_bf16(pA, vb3, o3);
    __syncthreads();
  }
  // normalize + write attn_vec (bf16, [B,T,H*D])
#pragma unroll
  for (int g = 0; g < 8; ++g) {
    const int i = i0 + mh + g;
    const float inv = 1.0f / s_l[mh + g];
    __bf16* dst = attn + ((size_t)b * TT + i) * HD + h * DD;
    dst[ 0 + nl] = (__bf16)(o0[g] * inv);
    dst[16 + nl] = (__bf16)(o1[g] * inv);
    dst[32 + nl] = (__bf16)(o2[g] * inv);
    dst[48 + nl] = (__bf16)(o3[g] * inv);
  }
}

// ---------------- output projection: [B*T,HD] @ [HD,E] -> f32 out ----------------
__global__ void __launch_bounds__(32) k_out(const __bf16* __restrict__ A, const __bf16* __restrict__ WT,
                                            float* __restrict__ out) {
  const int n0 = blockIdx.x * 64;   // over E
  const int m0 = blockIdx.y * 16;   // over B*T
  v8f a0 = zero8(), a1 = zero8(), a2 = zero8(), a3 = zero8();
  gemm16x64(A + (size_t)m0 * HD, WT + (size_t)n0 * HD, HD, HD, HD, a0, a1, a2, a3);
  const int lane = laneid(), nl = lane & 15, mh = (lane >> 4) << 3;
  F8 u[4]; u[0].v = a0; u[1].v = a1; u[2].v = a2; u[3].v = a3;
#pragma unroll
  for (int nt = 0; nt < 4; ++nt) {
#pragma unroll
    for (int g = 0; g < 8; ++g) {
      const int m = m0 + mh + g;
      out[(size_t)m * EE + n0 + (nt << 4) + nl] = u[nt].f[g];
    }
  }
}

// ---------------- launcher ----------------
extern "C" void kernel_launch(void* const* d_in, const int* in_sizes, int n_in,
                              void* d_out, int out_size, void* d_ws, size_t ws_size,
                              hipStream_t stream) {
  const float* we   = (const float*)d_in[0];  // word_embedding [B,T,E]
  const float* pe   = (const float*)d_in[1];  // pos_enc [T,E]
  const float* wqkv = (const float*)d_in[2];  // [E,3HD]
  const float* wpos = (const float*)d_in[3];  // [E,HD]
  const float* wout = (const float*)d_in[4];  // [HD,E]
  const float* rrb  = (const float*)d_in[5];  // r_r_bias [H,D]
  const float* rwb  = (const float*)d_in[6];  // r_w_bias [H,D]
  const int*   lens = (const int*)d_in[7];    // [B]
  float* out = (float*)d_out;

  char* ws = (char*)d_ws;
  size_t off = 0;
  auto alloc = [&](size_t elems) -> __bf16* {
    __bf16* p = (__bf16*)(ws + off);
    off += ((elems * 2 + 255) / 256) * 256;
    return p;
  };
  __bf16* Xbf   = alloc((size_t)BB * TT * EE);
  __bf16* Pbf   = alloc((size_t)TT * EE);
  __bf16* WqkvT = alloc((size_t)N3 * EE);
  __bf16* WposT = alloc((size_t)HD * EE);
  __bf16* WoutT = alloc((size_t)EE * HD);
  __bf16* qwB   = alloc((size_t)BB * HH * TT * DD);
  __bf16* qrB   = alloc((size_t)BB * HH * TT * DD);
  __bf16* kkB   = alloc((size_t)BB * HH * TT * DD);
  __bf16* vTB   = alloc((size_t)BB * HH * DD * TT);
  __bf16* rrB   = alloc((size_t)HH * TT * DD);
  __bf16* atB   = alloc((size_t)BB * TT * HD);

  k_convert<<<dim3((BB * TT * EE + 255) / 256), 256, 0, stream>>>(we, Xbf, BB * TT * EE);
  k_convert<<<dim3((TT * EE + 255) / 256), 256, 0, stream>>>(pe, Pbf, TT * EE);
  k_transpose<<<dim3(N3 / 16, EE / 16), dim3(16, 16), 0, stream>>>(wqkv, WqkvT, EE, N3);
  k_transpose<<<dim3(HD / 16, EE / 16), dim3(16, 16), 0, stream>>>(wpos, WposT, EE, HD);
  k_transpose<<<dim3(EE / 16, HD / 16), dim3(16, 16), 0, stream>>>(wout, WoutT, HD, EE);
  k_qkv<<<dim3(N3 / 64, (BB * TT) / 16), 32, 0, stream>>>(Xbf, WqkvT, rwb, rrb, qwB, qrB, kkB, vTB);
  k_pos<<<dim3(HD / 64, TT / 16), 32, 0, stream>>>(Pbf, WposT, rrB);
  k_attn<<<dim3(TT / 16, HH, BB), 32, 0, stream>>>(qwB, qrB, kkB, vTB, rrB, lens, atB);
  k_out<<<dim3(EE / 64, (BB * TT) / 16), 32, 0, stream>>>(atB, WoutT, out);

  (void)in_sizes; (void)n_in; (void)out_size; (void)ws_size;
}
